// Attention_54932631716178
// MI455X (gfx1250) — matrix operand order repaired and checked by reading, not compiled, not run on hardware
//
#include <hip/hip_runtime.h>
#include <hip/hip_bf16.h>

// ---------------- problem constants ----------------
#define BB 2
#define LL 2048
#define DD 2048
#define HH 16
#define DH 128
#define MM (BB * LL)   // 4096 rows total across batch

typedef __attribute__((ext_vector_type(16))) __bf16 v16bf;
typedef __attribute__((ext_vector_type(8)))  float  v8f;

union BFrag { v16bf v; uint4 q[2]; __bf16 e[16]; };
union CFrag { v8f v; float f[8]; };

// ---------------- f32 -> bf16 cast ----------------
__global__ void cast_f32_to_bf16(const float* __restrict__ in,
                                 __bf16* __restrict__ out, size_t n) {
  size_t i = (size_t)blockIdx.x * blockDim.x + threadIdx.x;
  size_t stride = (size_t)gridDim.x * blockDim.x;
  for (; i < n; i += stride) out[i] = (__bf16)in[i];
}

// ---------------- GEMM core: one wave computes 32x64 of C = X * W^T ----------------
// X: [M x K] bf16 row-major.  W: [N x K] bf16 row-major (B-matrix columns = W rows,
// contiguous in memory).  2 A-frags x 4 B-frags -> 8 v_wmma per 32-deep k-step,
// 24 B/lane loaded per WMMA.
__device__ __forceinline__ void gemm_tile_32x64(const __bf16* __restrict__ X,
                                                const __bf16* __restrict__ W,
                                                int K, int m0, int n0, int lane,
                                                v8f acc[2][4]) {
  const int half = lane >> 4;
  const int l16  = lane & 15;
  const __bf16* xrow0 = X + (size_t)(m0 + l16) * K;
  const __bf16* xrow1 = X + (size_t)(m0 + 16 + l16) * K;
  for (int k0 = 0; k0 < K; k0 += 32) {
    // speculative prefetch of the A stream a few k-steps ahead (global_prefetch_b8)
    __builtin_prefetch((const void*)(xrow0 + k0 + 256), 0, 0);
    __builtin_prefetch((const void*)(xrow1 + k0 + 256), 0, 0);
    BFrag a0, a1;
    // A-frag (16x32): lane<16 -> K {0..7,16..23}; lane>=16 -> K {8..15,24..31}
    a0.q[0] = *(const uint4*)(xrow0 + k0 + 8 * half);
    a0.q[1] = *(const uint4*)(xrow0 + k0 + 16 + 8 * half);
    a1.q[0] = *(const uint4*)(xrow1 + k0 + 8 * half);
    a1.q[1] = *(const uint4*)(xrow1 + k0 + 16 + 8 * half);
#pragma unroll
    for (int nn = 0; nn < 4; ++nn) {
      const int ncol = n0 + nn * 16 + l16;
      const __bf16* wrow = W + (size_t)ncol * K + k0 + 8 * half;
      BFrag b;
      // B-frag (32x16): lane = column; lane<16 -> K {0..7,16..23}; lane>=16 -> K {8..15,24..31}
      b.q[0] = *(const uint4*)(wrow);
      b.q[1] = *(const uint4*)(wrow + 16);
      acc[0][nn] = __builtin_amdgcn_wmma_f32_16x16x32_bf16(
          false, a0.v, false, b.v, (short)0, acc[0][nn], false, false);
      acc[1][nn] = __builtin_amdgcn_wmma_f32_16x16x32_bf16(
          false, a1.v, false, b.v, (short)0, acc[1][nn], false, false);
    }
  }
}

// ---------------- projection into head-split [B,H,L,DH] bf16 ----------------
__global__ __launch_bounds__(256) void proj_qkv_kernel(
    const __bf16* __restrict__ X, const __bf16* __restrict__ W,
    const float* __restrict__ bias, __bf16* __restrict__ out_hs) {
  const int gw = blockIdx.x * (blockDim.x >> 5) + (threadIdx.x >> 5);
  const int tiles_n = DD / 64;                 // 32
  const int mt = gw / tiles_n, ng = gw % tiles_n;
  if (mt >= MM / 32) return;
  const int lane = threadIdx.x & 31, half = lane >> 4, l16 = lane & 15;
  const int m0 = mt * 32, n0 = ng * 64;

  v8f acc[2][4];
#pragma unroll
  for (int i = 0; i < 2; ++i)
#pragma unroll
    for (int j = 0; j < 4; ++j)
      acc[i][j] = (v8f){0.f,0.f,0.f,0.f,0.f,0.f,0.f,0.f};

  gemm_tile_32x64(X, W, DD, m0, n0, lane, acc);

#pragma unroll
  for (int mi = 0; mi < 2; ++mi) {
#pragma unroll
    for (int nn = 0; nn < 4; ++nn) {
      const int col = n0 + nn * 16 + l16;
      const int h = col >> 7, dh = col & (DH - 1);
      const float bv = bias[col];
      CFrag c; c.v = acc[mi][nn];
#pragma unroll
      for (int r = 0; r < 8; ++r) {
        const int row = m0 + mi * 16 + r + 8 * half;  // C layout: upper half = rows +8
        const int b_ = row >> 11, l = row & (LL - 1);
        out_hs[(((size_t)(b_ * HH + h)) * LL + l) * DH + dh] = (__bf16)(c.f[r] + bv);
      }
    }
  }
}

// ---------------- final projection: f32 flat [B,L,D] output ----------------
__global__ __launch_bounds__(256) void proj_out_kernel(
    const __bf16* __restrict__ X, const __bf16* __restrict__ W,
    const float* __restrict__ bias, float* __restrict__ out) {
  const int gw = blockIdx.x * (blockDim.x >> 5) + (threadIdx.x >> 5);
  const int tiles_n = DD / 64;
  const int mt = gw / tiles_n, ng = gw % tiles_n;
  if (mt >= MM / 32) return;
  const int lane = threadIdx.x & 31, half = lane >> 4, l16 = lane & 15;
  const int m0 = mt * 32, n0 = ng * 64;

  v8f acc[2][4];
#pragma unroll
  for (int i = 0; i < 2; ++i)
#pragma unroll
    for (int j = 0; j < 4; ++j)
      acc[i][j] = (v8f){0.f,0.f,0.f,0.f,0.f,0.f,0.f,0.f};

  gemm_tile_32x64(X, W, DD, m0, n0, lane, acc);

#pragma unroll
  for (int mi = 0; mi < 2; ++mi) {
#pragma unroll
    for (int nn = 0; nn < 4; ++nn) {
      const int col = n0 + nn * 16 + l16;
      const float bv = bias[col];
      CFrag c; c.v = acc[mi][nn];
#pragma unroll
      for (int r = 0; r < 8; ++r) {
        const int row = m0 + mi * 16 + r + 8 * half;
        out[(size_t)row * DD + col] = c.f[r] + bv;
      }
    }
  }
}

// ---------------- flash attention (causal, online softmax) ----------------
// One wave per 16-row Q block.  Score scale = sqrt(DH) (faithful to reference).
// V tiles staged LDS-direct via global_load_async_to_lds_b128 (ASYNCcnt path).
#define FA_WAVES 4
__global__ __launch_bounds__(128) void flash_attn_kernel(
    const __bf16* __restrict__ qh, const __bf16* __restrict__ kh,
    const __bf16* __restrict__ vh, __bf16* __restrict__ outb) {
  __shared__ __bf16 ldsV[FA_WAVES][32 * DH];   // V tile, row-major [j=32][dh=128]
  __shared__ __bf16 ldsP[FA_WAVES][16 * 32];   // P tile [m=16][j=32]

  const int w = threadIdx.x >> 5;
  const int gw = blockIdx.x * FA_WAVES + w;
  const int MT = LL / 16;                      // 128 m-tiles
  if (gw >= BB * HH * MT) return;
  const int mt = gw % MT;
  const int h  = (gw / MT) % HH;
  const int b  = gw / (MT * HH);
  const int lane = threadIdx.x & 31, half = lane >> 4, l16 = lane & 15;
  const int m0 = mt * 16;

  const __bf16* Qb = qh + ((size_t)(b * HH + h)) * LL * DH;
  const __bf16* Kb = kh + ((size_t)(b * HH + h)) * LL * DH;
  const __bf16* Vb = vh + ((size_t)(b * HH + h)) * LL * DH;

  // ---- preload Q fragments (16 rows x 128 K, as 4 A-frags of K=32) ----
  BFrag qf[4];
  {
    const __bf16* qrow = Qb + (size_t)(m0 + l16) * DH;
#pragma unroll
    for (int f = 0; f < 4; ++f) {
      qf[f].q[0] = *(const uint4*)(qrow + f * 32 + 8 * half);
      qf[f].q[1] = *(const uint4*)(qrow + f * 32 + 16 + 8 * half);
    }
  }

  CFrag o[8];
  float mm[8], ll[8];
#pragma unroll
  for (int g = 0; g < 8; ++g)
#pragma unroll
    for (int r = 0; r < 8; ++r) o[g].f[r] = 0.0f;
#pragma unroll
  for (int r = 0; r < 8; ++r) { mm[r] = -3.0e38f; ll[r] = 0.0f; }

  const float sscale = 11.313708499f;          // sqrt(128): reference's d**0.5 scale

  const int jtiles = (m0 + 16 + 31) / 32;      // causal extent in 32-key tiles
  for (int jt = 0; jt < jtiles; ++jt) {
    const int j0 = jt * 32;

    // ---- async-stage V tile [32 x 128] straight into LDS (no VGPR round trip) ----
    asm volatile("s_wait_dscnt 0x0" ::: "memory");   // prior-iter LDS reads done
#pragma unroll
    for (int i = 0; i < 16; ++i) {
      const int idx = i * 32 + lane;           // 0..511 16B-chunks, coalesced
      const int row = idx >> 4;                // 0..31
      const int ch  = idx & 15;
      unsigned lds_off = (unsigned)(size_t)(&ldsV[w][row * DH + ch * 8]);
      unsigned long long ga =
          (unsigned long long)(size_t)(Vb + (size_t)(j0 + row) * DH + ch * 8);
      asm volatile("global_load_async_to_lds_b128 %0, %1, off"
                   :: "v"(lds_off), "v"(ga) : "memory");
    }

    // ---- S = Q K^T over two 16-col sub-tiles (overlaps with async V copy) ----
    CFrag s[2];
#pragma unroll
    for (int sub = 0; sub < 2; ++sub) {
#pragma unroll
      for (int r = 0; r < 8; ++r) s[sub].f[r] = 0.0f;
      if (j0 + sub * 16 > m0 + 15) continue;   // wave-uniform: fully-masked sub-tile
      const int j = j0 + sub * 16 + l16;       // this lane's key/column
      const __bf16* krow = Kb + (size_t)j * DH;
#pragma unroll
      for (int f = 0; f < 4; ++f) {
        BFrag bk;                              // B columns = K rows (contiguous)
        bk.q[0] = *(const uint4*)(krow + f * 32 + 8 * half);
        bk.q[1] = *(const uint4*)(krow + f * 32 + 8 * half + 16);
        s[sub].v = __builtin_amdgcn_wmma_f32_16x16x32_bf16(
            false, qf[f].v, false, bk.v, (short)0, s[sub].v, false, false);
      }
    }

    // ---- scale + causal mask ----
#pragma unroll
    for (int sub = 0; sub < 2; ++sub) {
      const int j = j0 + sub * 16 + l16;
#pragma unroll
      for (int r = 0; r < 8; ++r) {
        const int row = m0 + r + 8 * half;
        const float x = s[sub].f[r] * sscale;
        s[sub].f[r] = (j > row) ? -1.0e30f : x;
      }
    }

    // ---- online softmax: row reductions across 16-lane groups ----
    float alpha[8];
#pragma unroll
    for (int r = 0; r < 8; ++r) {
      float mx = fmaxf(s[0].f[r], s[1].f[r]);
      mx = fmaxf(mx, __shfl_xor(mx, 1, 32));
      mx = fmaxf(mx, __shfl_xor(mx, 2, 32));
      mx = fmaxf(mx, __shfl_xor(mx, 4, 32));
      mx = fmaxf(mx, __shfl_xor(mx, 8, 32));
      const float mnew = fmaxf(mm[r], mx);
      alpha[r] = __expf(mm[r] - mnew);
      mm[r] = mnew;
      const float p0 = __expf(s[0].f[r] - mnew);
      const float p1 = __expf(s[1].f[r] - mnew);
      s[0].f[r] = p0; s[1].f[r] = p1;
      float ps = p0 + p1;
      ps += __shfl_xor(ps, 1, 32);
      ps += __shfl_xor(ps, 2, 32);
      ps += __shfl_xor(ps, 4, 32);
      ps += __shfl_xor(ps, 8, 32);
      ll[r] = ll[r] * alpha[r] + ps;
    }

    // ---- C-layout -> A-layout transpose of P through LDS ----
#pragma unroll
    for (int sub = 0; sub < 2; ++sub)
#pragma unroll
      for (int r = 0; r < 8; ++r)
        ldsP[w][(r + 8 * half) * 32 + sub * 16 + l16] = (__bf16)s[sub].f[r];
    asm volatile("s_wait_dscnt 0x0" ::: "memory");

    BFrag pf;                                  // P as 16x32 A-fragment
    pf.q[0] = *(const uint4*)(&ldsP[w][l16 * 32 + 8 * half]);
    pf.q[1] = *(const uint4*)(&ldsP[w][l16 * 32 + 16 + 8 * half]);

    // ---- wait for async V tile, then O = alpha*O + P V ----
    asm volatile("s_wait_asynccnt 0x0" ::: "memory");
#pragma unroll
    for (int g = 0; g < 8; ++g) {
#pragma unroll
      for (int r = 0; r < 8; ++r) o[g].f[r] *= alpha[r];
      BFrag bv;                                // B col = V column (gather from LDS)
#pragma unroll
      for (int t = 0; t < 16; ++t)
        bv.e[t] = ldsV[w][(8 * half + t + (t & 8)) * DH + g * 16 + l16];
      o[g].v = __builtin_amdgcn_wmma_f32_16x16x32_bf16(
          false, pf.v, false, bv.v, (short)0, o[g].v, false, false);
    }
  }

  // ---- normalize and store to [B,L,D] bf16 ----
  float invl[8];
#pragma unroll
  for (int r = 0; r < 8; ++r) invl[r] = 1.0f / ll[r];
#pragma unroll
  for (int g = 0; g < 8; ++g) {
#pragma unroll
    for (int r = 0; r < 8; ++r) {
      const int row = m0 + r + 8 * half;
      outb[((size_t)b * LL + row) * DD + h * DH + g * 16 + l16] =
          (__bf16)(o[g].f[r] * invl[r]);
    }
  }
}

// ---------------- host launcher ----------------
extern "C" void kernel_launch(void* const* d_in, const int* in_sizes, int n_in,
                              void* d_out, int out_size, void* d_ws, size_t ws_size,
                              hipStream_t stream) {
  (void)in_sizes; (void)n_in; (void)out_size; (void)ws_size;
  const float* q  = (const float*)d_in[0];
  const float* k  = (const float*)d_in[1];
  const float* v  = (const float*)d_in[2];
  const float* Wq = (const float*)d_in[3];
  const float* bq = (const float*)d_in[4];
  const float* Wk = (const float*)d_in[5];
  const float* bk = (const float*)d_in[6];
  const float* Wv = (const float*)d_in[7];
  const float* bv = (const float*)d_in[8];
  const float* Wo = (const float*)d_in[9];
  const float* bo = (const float*)d_in[10];
  float* out = (float*)d_out;

  const size_t S1 = (size_t)BB * LL * DD;      // 8,388,608 elems
  const size_t Wn = (size_t)DD * DD;           // 4,194,304 elems
  __bf16* qb  = (__bf16*)d_ws;
  __bf16* kb  = qb  + S1;
  __bf16* vb  = kb  + S1;
  __bf16* Wqb = vb  + S1;
  __bf16* Wkb = Wqb + Wn;
  __bf16* Wvb = Wkb + Wn;
  __bf16* Wob = Wvb + Wn;
  __bf16* qhb = Wob + Wn;                      // head-split [B,H,L,DH]
  __bf16* khb = qhb + S1;
  __bf16* vhb = khb + S1;
  __bf16* ab  = vhb + S1;                      // attention out [B,L,D]

  // 1) casts
  cast_f32_to_bf16<<<2048, 256, 0, stream>>>(q,  qb,  S1);
  cast_f32_to_bf16<<<2048, 256, 0, stream>>>(k,  kb,  S1);
  cast_f32_to_bf16<<<2048, 256, 0, stream>>>(v,  vb,  S1);
  cast_f32_to_bf16<<<1024, 256, 0, stream>>>(Wq, Wqb, Wn);
  cast_f32_to_bf16<<<1024, 256, 0, stream>>>(Wk, Wkb, Wn);
  cast_f32_to_bf16<<<1024, 256, 0, stream>>>(Wv, Wvb, Wn);
  cast_f32_to_bf16<<<1024, 256, 0, stream>>>(Wo, Wob, Wn);

  // 2) projections: (M/32)*(D/64) = 128*32 = 4096 wave-tiles / 8 waves per block
  const int proj_blocks = (MM / 32) * (DD / 64) / 8;   // 512
  proj_qkv_kernel<<<proj_blocks, 256, 0, stream>>>(qb, Wqb, bq, qhb);
  proj_qkv_kernel<<<proj_blocks, 256, 0, stream>>>(kb, Wkb, bk, khb);
  proj_qkv_kernel<<<proj_blocks, 256, 0, stream>>>(vb, Wvb, bv, vhb);

  // 3) flash attention: B*H*(L/16) = 4096 waves / 4 waves per block
  const int fa_blocks = BB * HH * (LL / 16) / FA_WAVES; // 1024
  flash_attn_kernel<<<fa_blocks, 128, 0, stream>>>(qhb, khb, vhb, ab);

  // 4) output projection -> f32 d_out
  proj_out_kernel<<<proj_blocks, 256, 0, stream>>>(ab, Wob, bo, out);
}
